// RNN_Variational_Encoder_56135222559310
// MI455X (gfx1250) — compile-verified, matrix operand order checked by reading, not run on hardware
//
#include <hip/hip_runtime.h>

// ---------------------------------------------------------------------------
// CDNA5 (gfx1250) ESN encoder: bf16 WMMA GEMMs + persistent grid-synced scan.
// ---------------------------------------------------------------------------

typedef __attribute__((ext_vector_type(16))) __bf16 v16bf;
typedef __attribute__((ext_vector_type(8)))  float  v8f;

#define ESN_B   64
#define ESN_T   512
#define ESN_D   80
#define ESN_DP  96      // D padded to multiple of 32
#define ESN_H   512
#define ESN_MH  1024
#define ESN_P   256

__device__ __forceinline__ __bf16 f2bf(float f) {
    union { float f; unsigned u; } v; v.f = f;
    unsigned r = v.u + 0x7FFFu + ((v.u >> 16) & 1u);   // RNE
    union { unsigned short s; __bf16 b; } o; o.s = (unsigned short)(r >> 16);
    return o.b;
}
__device__ __forceinline__ float bf2f(__bf16 b) {
    union { unsigned short s; __bf16 b; } i; i.b = b;
    union { unsigned u; float f; } o; o.u = ((unsigned)i.s) << 16;
    return o.f;
}

// Load one 16x32 (MxK or NxK) bf16 WMMA operand fragment.
// Source is row-major [rows, ld], k-contiguous. Per ISA 7.12.2 layout:
// lane L holds row (base + L%16); K values { kh..kh+7 , kh+16..kh+23 }, kh=8*(L>=16).
__device__ __forceinline__ v16bf load_frag(const __bf16* __restrict__ src,
                                           int row0, int maxrow, int ld, int kb) {
    int lane = threadIdx.x & 31;
    int r = row0 + (lane & 15);
    if (r > maxrow) r = maxrow;
    const __bf16* p = src + (size_t)r * ld + kb + ((lane >> 4) << 3);
    v16bf f;
    ((uint4*)&f)[0] = *(const uint4*)(p);        // K = kh .. kh+7
    ((uint4*)&f)[1] = *(const uint4*)(p + 16);   // K = kh+16 .. kh+23
    return f;
}

__device__ __forceinline__ v8f bwmma(v16bf a, v16bf b, v8f c) {
    return __builtin_amdgcn_wmma_f32_16x16x32_bf16(false, a, false, b,
                                                   (short)0, c, false, false);
}

// ---------------------------------------------------------------------------
// fp32 -> bf16 convert with optional column zero-padding (rows x icols -> rows x ocols)
// ---------------------------------------------------------------------------
__global__ void cvt_pad(const float* __restrict__ src, __bf16* __restrict__ dst,
                        int rows, int icols, int ocols) {
    long i = (long)blockIdx.x * blockDim.x + threadIdx.x;
    long n = (long)rows * ocols;
    if (i >= n) return;
    int r = (int)(i / ocols), c = (int)(i % ocols);
    float v = (c < icols) ? src[(long)r * icols + c] : 0.0f;
    dst[i] = f2bf(v);
}

__global__ void zero_ws(unsigned* __restrict__ ctr, unsigned* __restrict__ hbuf, int nwords) {
    int i = blockIdx.x * blockDim.x + threadIdx.x;
    if (i < 4) ctr[i] = 0u;
    for (int k = i; k < nwords; k += gridDim.x * blockDim.x) hbuf[k] = 0u;
}

// ---------------------------------------------------------------------------
// Generic WMMA GEMM: C[M,N] = act( [A0|A1][M,K] * Bw[N,K]^T + bias )
// A is optionally K-concatenated from two row-major buffers (split at kSplit).
// Block: 256 threads = 8 waves; tile 128(M) x 64(N); each wave 16x64.
// ---------------------------------------------------------------------------
__global__ void wmma_gemm(const __bf16* __restrict__ A0, const __bf16* __restrict__ A1,
                          int kSplit, int M, int N, int K,
                          const __bf16* __restrict__ Bw,
                          const float* __restrict__ bias, int act_tanh,
                          __bf16* __restrict__ Co, float* __restrict__ Cf) {
    int wave = threadIdx.x >> 5, lane = threadIdx.x & 31;
    int m0 = blockIdx.y * 128 + wave * 16;
    int n0 = blockIdx.x * 64;

    v8f zero = {0.f,0.f,0.f,0.f,0.f,0.f,0.f,0.f};
    v8f acc[4] = {zero, zero, zero, zero};

    for (int kb = 0; kb < K; kb += 32) {
        const __bf16* Ap = A0; int kk = kb; int lda = kSplit;
        if (kb >= kSplit) { Ap = A1; kk = kb - kSplit; lda = K - kSplit; }
        v16bf a = load_frag(Ap, m0, M - 1, lda, kk);
#pragma unroll
        for (int j = 0; j < 4; ++j) {
            v16bf b = load_frag(Bw, n0 + 16 * j, N - 1, K, kb);
            acc[j] = bwmma(a, b, acc[j]);
        }
    }

    int colL = lane & 15, rhi = (lane >> 4) << 3;
#pragma unroll
    for (int j = 0; j < 4; ++j) {
#pragma unroll
        for (int r = 0; r < 8; ++r) {
            int m = m0 + r + rhi;
            if (m >= M) continue;
            int n = n0 + 16 * j + colL;
            float v = acc[j][r];
            if (bias) v += bias[n];
            if (act_tanh) v = tanhf(v);
            size_t idx = (size_t)m * N + n;
            if (Co) Co[idx] = f2bf(v);
            else    Cf[idx] = v;
        }
    }
}

// ---------------------------------------------------------------------------
// Persistent ESN scan (both directions of one layer).
// 16 blocks: dir = blk/8, 8 blocks per direction each owning 64 columns of H.
// Per step: h_new = tanh(i2h[t] + h * W_hh^T), masked by length; h ping-pongs
// in global (L2-resident); timestep barrier = device-scope atomic rel/acq.
// i2h / hseq layout: [B, T, H] (row = b*T + t). hbuf: [2][B*H] bf16.
// ---------------------------------------------------------------------------
__global__ void esn_scan(const __bf16* __restrict__ i2h0, const __bf16* __restrict__ i2h1,
                         const __bf16* __restrict__ whh0, const __bf16* __restrict__ whh1,
                         __bf16* __restrict__ hseq0, __bf16* __restrict__ hseq1,
                         __bf16* __restrict__ hbuf0, __bf16* __restrict__ hbuf1,
                         const int* __restrict__ lengths, unsigned* __restrict__ ctr, int T) {
    const int H = ESN_H, BH = ESN_B * ESN_H;
    int dir = blockIdx.x >> 3;
    int nb  = blockIdx.x & 7;
    const __bf16* i2h  = dir ? i2h1  : i2h0;
    const __bf16* whh  = dir ? whh1  : whh0;
    __bf16*       hseq = dir ? hseq1 : hseq0;
    __bf16*       hbuf = dir ? hbuf1 : hbuf0;
    unsigned*     c    = ctr + dir;

    int wave = threadIdx.x >> 5, lane = threadIdx.x & 31;
    int m0 = (wave & 3) * 16;                 // batch rows
    int n0 = nb * 64 + (wave >> 2) * 32;      // hidden cols (2 wmma tiles)
    int colL = lane & 15, rhi = (lane >> 4) << 3;
    v8f zero = {0.f,0.f,0.f,0.f,0.f,0.f,0.f,0.f};

    for (int t = 0; t < T; ++t) {
        const __bf16* cur = hbuf + (t & 1) * BH;
        __bf16*       nxt = hbuf + ((t + 1) & 1) * BH;

        v8f acc0 = zero, acc1 = zero;
        for (int kb = 0; kb < H; kb += 32) {
            v16bf a  = load_frag(cur, m0, ESN_B - 1, H, kb);
            v16bf b0 = load_frag(whh, n0,      H - 1, H, kb);
            v16bf b1 = load_frag(whh, n0 + 16, H - 1, H, kb);
            acc0 = bwmma(a, b0, acc0);
            acc1 = bwmma(a, b1, acc1);
        }

#pragma unroll
        for (int r = 0; r < 8; ++r) {
            int m = m0 + r + rhi;
            bool msk = t < lengths[m];
            size_t rbt = ((size_t)m * T + t) * H;
            {
                int n = n0 + colL;
                float v = tanhf(acc0[r] + bf2f(i2h[rbt + n]));
                __bf16 hv = msk ? f2bf(v) : cur[m * H + n];
                nxt[m * H + n] = hv; hseq[rbt + n] = hv;
            }
            {
                int n = n0 + 16 + colL;
                float v = tanhf(acc1[r] + bf2f(i2h[rbt + n]));
                __bf16 hv = msk ? f2bf(v) : cur[m * H + n];
                nxt[m * H + n] = hv; hseq[rbt + n] = hv;
            }
        }

        __threadfence();
        __syncthreads();
        if (threadIdx.x == 0) {
            __hip_atomic_fetch_add(c, 1u, __ATOMIC_RELEASE, __HIP_MEMORY_SCOPE_AGENT);
            unsigned tgt = 8u * (unsigned)(t + 1);
            while (__hip_atomic_load(c, __ATOMIC_ACQUIRE, __HIP_MEMORY_SCOPE_AGENT) < tgt)
                __builtin_amdgcn_s_sleep(2);
        }
        __syncthreads();
    }
}

// features[64, 4H] = concat(h_l0f, h_l0b, h_l1f, h_l1b) final states (buf 0, T even)
__global__ void build_feat(const __bf16* __restrict__ h0f, const __bf16* __restrict__ h0b,
                           const __bf16* __restrict__ h1f, const __bf16* __restrict__ h1b,
                           __bf16* __restrict__ feat) {
    int i = blockIdx.x * blockDim.x + threadIdx.x;
    if (i >= ESN_B * 4 * ESN_H) return;
    int b = i >> 11, c = i & 2047;
    const __bf16* s = (c < 512) ? h0f : (c < 1024) ? h0b : (c < 1536) ? h1f : h1b;
    feat[i] = s[b * ESN_H + (c & 511)];
}

// ---------------------------------------------------------------------------
extern "C" void kernel_launch(void* const* d_in, const int* in_sizes, int n_in,
                              void* d_out, int out_size, void* d_ws, size_t ws_size,
                              hipStream_t stream) {
    const float* x        = (const float*)d_in[0];
    const int*   lengths  = (const int*)d_in[1];
    const float* w_ih_l0  = (const float*)d_in[2];
    const float* w_hh_l0  = (const float*)d_in[3];
    const float* w_ih_l0r = (const float*)d_in[4];
    const float* w_hh_l0r = (const float*)d_in[5];
    const float* w_ih_l1  = (const float*)d_in[6];
    const float* w_hh_l1  = (const float*)d_in[7];
    const float* w_ih_l1r = (const float*)d_in[8];
    const float* w_hh_l1r = (const float*)d_in[9];
    const float* mw1[2] = {(const float*)d_in[10], (const float*)d_in[14]};
    const float* mb1[2] = {(const float*)d_in[11], (const float*)d_in[15]};
    const float* mw2[2] = {(const float*)d_in[12], (const float*)d_in[16]};
    const float* mb2[2] = {(const float*)d_in[13], (const float*)d_in[17]};
    float* out = (float*)d_out;

    char* ws = (char*)d_ws;
    size_t off = 0;
    auto alloc = [&](size_t bytes) -> char* {
        char* p = ws + off;
        off = (off + bytes + 255) & ~(size_t)255;
        return p;
    };
    const int BT = ESN_B * ESN_T;                  // 32768
    unsigned* ctr   = (unsigned*)alloc(256);
    __bf16* xpad    = (__bf16*)alloc((size_t)BT * ESN_DP * 2);
    __bf16* wih0f   = (__bf16*)alloc((size_t)ESN_H * ESN_DP * 2);
    __bf16* wih0b   = (__bf16*)alloc((size_t)ESN_H * ESN_DP * 2);
    __bf16* whh0f   = (__bf16*)alloc((size_t)ESN_H * ESN_H * 2);
    __bf16* whh0b   = (__bf16*)alloc((size_t)ESN_H * ESN_H * 2);
    __bf16* wih1f   = (__bf16*)alloc((size_t)ESN_H * 2 * ESN_H * 2);
    __bf16* wih1b   = (__bf16*)alloc((size_t)ESN_H * 2 * ESN_H * 2);
    __bf16* whh1f   = (__bf16*)alloc((size_t)ESN_H * ESN_H * 2);
    __bf16* whh1b   = (__bf16*)alloc((size_t)ESN_H * ESN_H * 2);
    __bf16* w1c[2], *w2c[2];
    for (int h = 0; h < 2; ++h) {
        w1c[h] = (__bf16*)alloc((size_t)ESN_MH * 4 * ESN_H * 2);
        w2c[h] = (__bf16*)alloc((size_t)ESN_P * ESN_MH * 2);
    }
    __bf16* i2hf = (__bf16*)alloc((size_t)BT * ESN_H * 2);   // reused for layer 1
    __bf16* i2hb = (__bf16*)alloc((size_t)BT * ESN_H * 2);
    __bf16* hsf  = (__bf16*)alloc((size_t)BT * ESN_H * 2);
    __bf16* hsb  = (__bf16*)alloc((size_t)BT * ESN_H * 2);
    __bf16* hbuf = (__bf16*)alloc((size_t)8 * ESN_B * ESN_H * 2);  // 4 dirs x 2 bufs
    __bf16* feat = (__bf16*)alloc((size_t)ESN_B * 4 * ESN_H * 2);
    __bf16* hid[2];
    hid[0] = (__bf16*)alloc((size_t)ESN_B * ESN_MH * 2);
    hid[1] = (__bf16*)alloc((size_t)ESN_B * ESN_MH * 2);

    const int BH = ESN_B * ESN_H;
    __bf16* hb0f = hbuf;            // layer0 fwd  [2][B*H]
    __bf16* hb0b = hbuf + 2 * BH;   // layer0 rev
    __bf16* hb1f = hbuf + 4 * BH;   // layer1 fwd
    __bf16* hb1b = hbuf + 6 * BH;   // layer1 rev

    // --- init: zero barrier counters + all h ping-pong buffers ---
    zero_ws<<<64, 256, 0, stream>>>(ctr, (unsigned*)hbuf, 8 * BH / 2);

    // --- convert inputs/weights to bf16 (pad K of layer-0 i2h to 96) ---
    auto cvt = [&](const float* s, __bf16* d, int r, int ic, int oc) {
        long n = (long)r * oc;
        cvt_pad<<<(unsigned)((n + 255) / 256), 256, 0, stream>>>(s, d, r, ic, oc);
    };
    cvt(x, xpad, BT, ESN_D, ESN_DP);
    cvt(w_ih_l0,  wih0f, ESN_H, ESN_D, ESN_DP);
    cvt(w_ih_l0r, wih0b, ESN_H, ESN_D, ESN_DP);
    cvt(w_hh_l0,  whh0f, ESN_H, ESN_H, ESN_H);
    cvt(w_hh_l0r, whh0b, ESN_H, ESN_H, ESN_H);
    cvt(w_ih_l1,  wih1f, ESN_H, 2 * ESN_H, 2 * ESN_H);
    cvt(w_ih_l1r, wih1b, ESN_H, 2 * ESN_H, 2 * ESN_H);
    cvt(w_hh_l1,  whh1f, ESN_H, ESN_H, ESN_H);
    cvt(w_hh_l1r, whh1b, ESN_H, ESN_H, ESN_H);
    for (int h = 0; h < 2; ++h) {
        cvt(mw1[h], w1c[h], ESN_MH, 4 * ESN_H, 4 * ESN_H);
        cvt(mw2[h], w2c[h], ESN_P, ESN_MH, ESN_MH);
    }

    // --- layer 0 input projections: i2h = Xpad * W_ih^T  ([B*T,96]x[96,512]) ---
    dim3 gBig(ESN_H / 64, (BT + 127) / 128);
    wmma_gemm<<<gBig, 256, 0, stream>>>(xpad, nullptr, ESN_DP, BT, ESN_H, ESN_DP,
                                        wih0f, nullptr, 0, i2hf, nullptr);
    wmma_gemm<<<gBig, 256, 0, stream>>>(xpad, nullptr, ESN_DP, BT, ESN_H, ESN_DP,
                                        wih0b, nullptr, 0, i2hb, nullptr);

    // --- layer 0 recurrent scan (both dirs, persistent, grid-synced) ---
    esn_scan<<<16, 256, 0, stream>>>(i2hf, i2hb, whh0f, whh0b, hsf, hsb,
                                     hb0f, hb0b, lengths, ctr, ESN_T);

    // --- layer 1 input projections: K-split A = [hseq_f | hseq_b], K=1024 ---
    wmma_gemm<<<gBig, 256, 0, stream>>>(hsf, hsb, ESN_H, BT, ESN_H, 2 * ESN_H,
                                        wih1f, nullptr, 0, i2hf, nullptr);
    wmma_gemm<<<gBig, 256, 0, stream>>>(hsf, hsb, ESN_H, BT, ESN_H, 2 * ESN_H,
                                        wih1b, nullptr, 0, i2hb, nullptr);

    // --- layer 1 recurrent scan (hseq buffers reused as scratch sink) ---
    esn_scan<<<16, 256, 0, stream>>>(i2hf, i2hb, whh1f, whh1b, hsf, hsb,
                                     hb1f, hb1b, lengths, ctr + 2, ESN_T);

    // --- features [64, 2048] (T even -> final state lives in buffer 0) ---
    build_feat<<<(ESN_B * 4 * ESN_H + 255) / 256, 256, 0, stream>>>(hb0f, hb0b, hb1f, hb1b, feat);

    // --- MLP heads: tanh(feat*W1^T + b1) * W2^T + b2 ---
    for (int h = 0; h < 2; ++h) {
        dim3 g1(ESN_MH / 64, 1);
        wmma_gemm<<<g1, 256, 0, stream>>>(feat, nullptr, 4 * ESN_H, ESN_B, ESN_MH, 4 * ESN_H,
                                          w1c[h], mb1[h], 1, hid[h], nullptr);
        dim3 g2(ESN_P / 64, 1);
        wmma_gemm<<<g2, 256, 0, stream>>>(hid[h], nullptr, ESN_MH, ESN_B, ESN_P, ESN_MH,
                                          w2c[h], mb2[h], 0, nullptr,
                                          out + (size_t)h * ESN_B * ESN_P);
    }

    (void)in_sizes; (void)n_in; (void)out_size; (void)ws_size;
}
